// GVPHardGumbelPartitionerModel_51651276702449
// MI455X (gfx1250) — matrix operand order repaired and checked by reading, not compiled, not run on hardware
//
#include <hip/hip_runtime.h>
#include <hip/hip_bf16.h>

#define BB 16
#define NN 2048
#define DD 128
#define HH 64
#define KK (DD + HH)   // 192
#define KKP 208        // padded LDS row stride (multiple of 16 bf16 -> 32B-aligned frags)
#define SS 5           // MAX_CLUSTERS

typedef __attribute__((ext_vector_type(16))) __bf16 v16bf;
typedef __attribute__((ext_vector_type(8)))  float  v8f;

// ---------------------------------------------------------------- utilities
__device__ __forceinline__ float hash_uniform(int b, int n, int tag) {
    unsigned s = (unsigned)b * 2654435761u ^ (unsigned)n * 2246822519u ^
                 (unsigned)tag * 3266489917u ^ 0x9E3779B9u;
    s ^= s >> 16; s *= 0x7feb352du;
    s ^= s >> 15; s *= 0x846ca68bu;
    s ^= s >> 16;
    return ((float)(s >> 8) + 0.5f) * (1.0f / 16777216.0f);  // (0,1)
}

// ---------------------------------------------------------------- init
__global__ void init_kernel(const int* __restrict__ mask, int* __restrict__ avail,
                            float* __restrict__ hgru, float* __restrict__ outAdj) {
    int i = blockIdx.x * 256 + threadIdx.x;
    if (i < BB * NN) avail[i] = (mask[i] != 0) ? 1 : 0;
    if (i < BB * HH) hgru[i] = 0.0f;
    if (i < BB * SS * SS) {
        int rc = i % (SS * SS);
        outAdj[i] = ((rc / SS) == (rc % SS)) ? 0.0f : 1.0f;
    }
}

// one-shot W1 -> bf16 conversion, row-major [HH][KK]
__global__ void prep_w1_kernel(const float* __restrict__ W1,
                               unsigned short* __restrict__ w1bf) {
    int i = blockIdx.x * 256 + threadIdx.x;
    if (i < HH * KK) ((__bf16*)w1bf)[i] = static_cast<__bf16>(W1[i]);
}

// ---------------------------------------------------------------- context init
__global__ __launch_bounds__(128) void ctx_kernel(const float* __restrict__ x,
                                                  const float* __restrict__ Wc,
                                                  const float* __restrict__ bc,
                                                  float* __restrict__ ctx) {
    int b = blockIdx.x, d = threadIdx.x;
    __shared__ float meanv[DD];
    float acc = 0.0f;
    for (int n = 0; n < NN; ++n) acc += x[((size_t)b * NN + n) * DD + d];
    meanv[d] = acc * (1.0f / NN);
    __syncthreads();
    if (d < HH) {
        float s = bc[d];
        for (int k = 0; k < DD; ++k) s += Wc[d * DD + k] * meanv[k];
        ctx[b * HH + d] = s;
    }
}

// ---------------------------------------------------------------- WMMA MLP logits
// h = relu([x,ctx] @ W1.T + b1); logits = h @ W2.T + b2
__device__ __forceinline__ v8f wmma_tile(const __bf16* w1row, v16bf a, v8f c) {
    // lane's whole B fragment is 16 contiguous bf16 (32B, aligned) in w1sT[h][k0..k0+16)
    v16bf bf = *(const v16bf*)w1row;
    return __builtin_amdgcn_wmma_f32_16x16x32_bf16(false, a, false, bf, (short)0, c,
                                                   false, false);
}

__global__ __launch_bounds__(128) void logits_kernel(
    const float* __restrict__ x, const float* __restrict__ ctx,
    const unsigned short* __restrict__ w1bf, const float* __restrict__ b1,
    const float* __restrict__ W2, const float* __restrict__ b2,
    float* __restrict__ logits) {
    __shared__ __bf16 w1sT[HH][KKP];      // 26 KB   B-operand: w1sT[h][k] = W1[h][k]
    __shared__ float  hS[64][HH + 2];     // 16.5 KB hidden activations

    const int tid  = threadIdx.x;
    const int wave = tid >> 5;
    const int lane = tid & 31;
    const int M    = lane & 15;
    const int half = lane >> 4;

    // stage bf16 W1 with vector copies (no conversions in the hot kernel)
    for (int i = tid; i < (HH * KK) / 8; i += 128) {
        int e = i * 8;
        int h = e / KK, k = e % KK;            // KK divisible by 8
        *(uint4*)&w1sT[h][k] = *(const uint4*)&w1bf[e];
    }
    __syncthreads();

    const int row = blockIdx.x * 64 + wave * 16 + M;  // rows of a block stay in one batch
    const int b = row / NN;
    const float* xrow = x + (size_t)row * DD;
    const float* crow = ctx + b * HH;
    const int kbase = half ? 16 : 0;  // A 16x32: lanes 0-15 K 0-15, lanes 16-31 K 16-31
    const int cb = lane & 15;

    v8f c0 = {}, c1 = {}, c2 = {}, c3 = {};

#pragma unroll
    for (int kc = 0; kc < KK; kc += 32) {
        // direct global B128 loads of this lane's A-fragment (two runs of 8 floats)
        const float* src = (kc < DD) ? (xrow + kc) : (crow + (kc - DD));
        float4 lo0 = *(const float4*)(src + kbase);
        float4 lo1 = *(const float4*)(src + kbase + 4);
        float4 hi0 = *(const float4*)(src + kbase + 8);
        float4 hi1 = *(const float4*)(src + kbase + 12);

        v16bf a;
        a[0]  = static_cast<__bf16>(lo0.x); a[1]  = static_cast<__bf16>(lo0.y);
        a[2]  = static_cast<__bf16>(lo0.z); a[3]  = static_cast<__bf16>(lo0.w);
        a[4]  = static_cast<__bf16>(lo1.x); a[5]  = static_cast<__bf16>(lo1.y);
        a[6]  = static_cast<__bf16>(lo1.z); a[7]  = static_cast<__bf16>(lo1.w);
        a[8]  = static_cast<__bf16>(hi0.x); a[9]  = static_cast<__bf16>(hi0.y);
        a[10] = static_cast<__bf16>(hi0.z); a[11] = static_cast<__bf16>(hi0.w);
        a[12] = static_cast<__bf16>(hi1.x); a[13] = static_cast<__bf16>(hi1.y);
        a[14] = static_cast<__bf16>(hi1.z); a[15] = static_cast<__bf16>(hi1.w);

        const int k0 = kc + half * 16;         // B 32x16: lanes 0-15 K 0-15, lanes 16-31 K 16-31
        c0 = wmma_tile(&w1sT[ 0 + cb][k0], a, c0);
        c1 = wmma_tile(&w1sT[16 + cb][k0], a, c1);
        c2 = wmma_tile(&w1sT[32 + cb][k0], a, c2);
        c3 = wmma_tile(&w1sT[48 + cb][k0], a, c3);
    }

    // unpack D (VGPR j -> row j + 8*half, col = lane&15), bias + relu
#pragma unroll
    for (int j = 0; j < 8; ++j) {
        int rl = wave * 16 + j + 8 * half;
        hS[rl][cb +  0] = fmaxf(c0[j] + b1[cb +  0], 0.0f);
        hS[rl][cb + 16] = fmaxf(c1[j] + b1[cb + 16], 0.0f);
        hS[rl][cb + 32] = fmaxf(c2[j] + b1[cb + 32], 0.0f);
        hS[rl][cb + 48] = fmaxf(c3[j] + b1[cb + 48], 0.0f);
    }
    __syncthreads();

    if (tid < 64) {
        float s = b2[0];
        for (int h = 0; h < HH; ++h) s += hS[tid][h] * W2[h];
        logits[blockIdx.x * 64 + tid] = s;
    }
}

// ---------------------------------------------------------------- gumbel argmax select
__global__ __launch_bounds__(256) void select_kernel(
    const float* __restrict__ logits, int* __restrict__ avail,
    const int* __restrict__ reachR, float* __restrict__ member,
    int* __restrict__ cur, int* __restrict__ reachW, int tag, int mode) {
    int b = blockIdx.x;
    __shared__ float bv[256];
    __shared__ int   bi[256];
    float best = -3.0e38f; int bidx = -1;
    for (int n = threadIdx.x; n < NN; n += 256) {
        int idx = b * NN + n;
        if (mode == 0) { member[idx] = 0.0f; cur[idx] = 0; reachW[idx] = 0; }
        bool cand = (avail[idx] != 0) && (mode == 0 || reachR[idx] != 0);
        if (cand) {
            float u = hash_uniform(b, n, tag);
            float g = -logf(-logf(u + 1e-8f) + 1e-8f);
            float v = logits[idx] + g;
            if (v > best) { best = v; bidx = n; }
        }
    }
    bv[threadIdx.x] = best; bi[threadIdx.x] = bidx;
    __syncthreads();
    for (int s = 128; s > 0; s >>= 1) {
        if (threadIdx.x < s) {
            float ov = bv[threadIdx.x + s]; int oi = bi[threadIdx.x + s];
            if (oi >= 0 && (bi[threadIdx.x] < 0 || ov > bv[threadIdx.x] ||
                            (ov == bv[threadIdx.x] && oi < bi[threadIdx.x]))) {
                bv[threadIdx.x] = ov; bi[threadIdx.x] = oi;
            }
        }
        __syncthreads();
    }
    if (threadIdx.x == 0 && bi[0] >= 0) {
        int idx = b * NN + bi[0];
        avail[idx] = 0;
        member[idx] = 1.0f;
        if (mode == 0) { cur[idx] = 1; reachW[idx] = 1; }
    }
}

// ---------------------------------------------------------------- frontier-sparse BFS hop
__global__ __launch_bounds__(256) void bfs_kernel(const float* __restrict__ adj,
                                                  const int* __restrict__ cur,
                                                  int* __restrict__ curNext,
                                                  int* __restrict__ reach) {
    int b = blockIdx.y;
    int m = blockIdx.x * 256 + threadIdx.x;
    __shared__ int list[NN];
    __shared__ int cnt;
    if (threadIdx.x == 0) cnt = 0;
    __syncthreads();
    for (int n = threadIdx.x; n < NN; n += 256)
        if (cur[b * NN + n]) { int p = atomicAdd(&cnt, 1); list[p] = n; }
    __syncthreads();
    bool nb = false;
    const size_t base = (size_t)b * NN * NN;
    for (int i = 0; i < cnt; ++i) {
        if (i + 1 < cnt)
            __builtin_prefetch(&adj[base + (size_t)list[i + 1] * NN + m], 0, 0);
        if (adj[base + (size_t)list[i] * NN + m] != 0.0f) nb = true;
    }
    int idx = b * NN + m;
    int newf = (nb && !reach[idx]) ? 1 : 0;
    curNext[idx] = newf;
    if (nb) reach[idx] = 1;
}

// ---------------------------------------------------------------- member-mean embedding
__global__ __launch_bounds__(128) void embed_kernel(const float* __restrict__ x,
                                                    const float* __restrict__ member,
                                                    float* __restrict__ hist,
                                                    float* __restrict__ outFeat, int c) {
    int b = blockIdx.x;
    __shared__ int list[8];
    __shared__ int cnt;
    if (threadIdx.x == 0) cnt = 0;
    __syncthreads();
    for (int n = threadIdx.x; n < NN; n += 128)
        if (member[b * NN + n] > 0.5f) {
            int p = atomicAdd(&cnt, 1);
            if (p < 8) list[p] = n;
        }
    __syncthreads();
    int d = threadIdx.x;
    int k = cnt < 8 ? cnt : 8;
    float acc = 0.0f;
    for (int i = 0; i < k; ++i) acc += x[((size_t)b * NN + list[i]) * DD + d];
    float e = acc / fmaxf((float)cnt, 1.0f);
    hist[(b * SS + c) * DD + d]    = e;
    outFeat[(b * SS + c) * DD + d] = e;
}

__global__ void assign_kernel(const float* __restrict__ member,
                              float* __restrict__ outAsg, int c) {
    int i = blockIdx.x * 256 + threadIdx.x;
    if (i < BB * NN) outAsg[(size_t)i * SS + c] = member[i];
}

// ---------------------------------------------------------------- GRU re-run over history
__global__ __launch_bounds__(64) void gru_kernel(const float* __restrict__ hist,
                                                 float* __restrict__ hgru,
                                                 float* __restrict__ ctxOut,
                                                 const float* __restrict__ Wih,
                                                 const float* __restrict__ Whh,
                                                 const float* __restrict__ bih,
                                                 const float* __restrict__ bhh, int T) {
    int b = blockIdx.x, j = threadIdx.x;
    __shared__ float h[HH], hn[HH], xt[DD];
    h[j] = hgru[b * HH + j];
    __syncthreads();
    for (int t = 0; t < T; ++t) {
        xt[j]      = hist[(b * SS + t) * DD + j];
        xt[j + 64] = hist[(b * SS + t) * DD + j + 64];
        __syncthreads();
        float gir = bih[j], giz = bih[HH + j], gin = bih[2 * HH + j];
        for (int k = 0; k < DD; ++k) {
            float xv = xt[k];
            gir += Wih[j * DD + k] * xv;
            giz += Wih[(HH + j) * DD + k] * xv;
            gin += Wih[(2 * HH + j) * DD + k] * xv;
        }
        float ghr = bhh[j], ghz = bhh[HH + j], ghn = bhh[2 * HH + j];
        for (int k = 0; k < HH; ++k) {
            float hv = h[k];
            ghr += Whh[j * HH + k] * hv;
            ghz += Whh[(HH + j) * HH + k] * hv;
            ghn += Whh[(2 * HH + j) * HH + k] * hv;
        }
        float r  = 1.0f / (1.0f + expf(-(gir + ghr)));
        float z  = 1.0f / (1.0f + expf(-(giz + ghz)));
        float nw = tanhf(gin + r * ghn);
        hn[j] = (1.0f - z) * nw + z * h[j];
        __syncthreads();
        h[j] = hn[j];
        __syncthreads();
    }
    hgru[b * HH + j]   = h[j];
    ctxOut[b * HH + j] = h[j];
}

// ---------------------------------------------------------------- host orchestration
extern "C" void kernel_launch(void* const* d_in, const int* in_sizes, int n_in,
                              void* d_out, int out_size, void* d_ws, size_t ws_size,
                              hipStream_t stream) {
    const float* x   = (const float*)d_in[0];
    const float* adj = (const float*)d_in[1];
    const int*   mask= (const int*)d_in[2];
    const float* W1  = (const float*)d_in[3];
    const float* b1  = (const float*)d_in[4];
    const float* W2  = (const float*)d_in[5];
    const float* b2  = (const float*)d_in[6];
    const float* Wc  = (const float*)d_in[7];
    const float* bc  = (const float*)d_in[8];
    const float* Wih = (const float*)d_in[9];
    const float* Whh = (const float*)d_in[10];
    const float* bih = (const float*)d_in[11];
    const float* bhh = (const float*)d_in[12];

    float* out     = (float*)d_out;
    float* outFeat = out;                                 // [B,S,D]
    float* outAdj  = out + BB * SS * DD;                  // [B,S,S]
    float* outAsg  = out + BB * SS * DD + BB * SS * SS;   // [B,N,S]

    char* w = (char*)d_ws;
    float* ctx    = (float*)w; w += (size_t)BB * HH * 4;
    float* hgru   = (float*)w; w += (size_t)BB * HH * 4;
    float* logits = (float*)w; w += (size_t)BB * NN * 4;
    float* member = (float*)w; w += (size_t)BB * NN * 4;
    float* hist   = (float*)w; w += (size_t)BB * SS * DD * 4;
    int* avail = (int*)w; w += (size_t)BB * NN * 4;
    int* reach = (int*)w; w += (size_t)BB * NN * 4;
    int* curA  = (int*)w; w += (size_t)BB * NN * 4;
    int* curB  = (int*)w; w += (size_t)BB * NN * 4;
    unsigned short* w1bf = (unsigned short*)w; w += (size_t)HH * KK * 2;

    init_kernel<<<(BB * NN + 255) / 256, 256, 0, stream>>>(mask, avail, hgru, outAdj);
    prep_w1_kernel<<<(HH * KK + 255) / 256, 256, 0, stream>>>(W1, w1bf);
    ctx_kernel<<<BB, 128, 0, stream>>>(x, Wc, bc, ctx);

    for (int c = 0; c < SS; ++c) {
        logits_kernel<<<BB * NN / 64, 128, 0, stream>>>(x, ctx, w1bf, b1, W2, b2, logits);
        // seed selection (zeroes member/cur/reach for this cluster)
        select_kernel<<<BB, 256, 0, stream>>>(logits, avail, reach, member,
                                              curA, reach, 2 * c, 0);
        // 2-hop frontier-sparse BFS
        bfs_kernel<<<dim3(NN / 256, BB), 256, 0, stream>>>(adj, curA, curB, reach);
        bfs_kernel<<<dim3(NN / 256, BB), 256, 0, stream>>>(adj, curB, curA, reach);
        // two more member selections within reach
        select_kernel<<<BB, 256, 0, stream>>>(logits, avail, reach, member,
                                              curA, reach, 2 * c + 1000, 1);
        select_kernel<<<BB, 256, 0, stream>>>(logits, avail, reach, member,
                                              curA, reach, 2 * c + 1001, 1);
        embed_kernel<<<BB, 128, 0, stream>>>(x, member, hist, outFeat, c);
        assign_kernel<<<(BB * NN + 255) / 256, 256, 0, stream>>>(member, outAsg, c);
        gru_kernel<<<BB, 64, 0, stream>>>(hist, hgru, ctx, Wih, Whh, bih, bhh, c + 1);
    }
    (void)in_sizes; (void)n_in; (void)out_size; (void)ws_size;
}
